// Renderer_54932631715864
// MI455X (gfx1250) — matrix operand order repaired and checked by reading, not compiled, not run on hardware
//
#include <hip/hip_runtime.h>
#include <hip/hip_bf16.h>

typedef __attribute__((ext_vector_type(16))) _Float16 v16h;
typedef __attribute__((ext_vector_type(8)))  _Float16 v8h;
typedef __attribute__((ext_vector_type(4)))  _Float16 v4h;
typedef __attribute__((ext_vector_type(8)))  float    v8f;

#define NVIEW 50
#define C1 64
#define C2 128
#define C3 512     // = 16*32
#define CCH 16
#define DD 32
#define NS 64
#define NPTS (NS*64*64)   // 262144

static __device__ __forceinline__ v16h cat8(v8h lo, v8h hi) {
  return __builtin_shufflevector(lo, hi, 0,1,2,3,4,5,6,7,8,9,10,11,12,13,14,15);
}

// ---------------------------------------------------------------------------
// Kernel 0: weight conversion / transposition (f32 -> f16) + zero the volume.
//  conv weights -> [N][K] with K ordered (kh*3+kw)*Cin + ci
//  MLP weights  -> transposed [n][k] (zero-padded)
// ---------------------------------------------------------------------------
__global__ __launch_bounds__(256)
void prep_kernel(const float* __restrict__ w2, const float* __restrict__ w3,
                 const float* __restrict__ mw1, const float* __restrict__ mw2,
                 const float* __restrict__ mw3,
                 _Float16* __restrict__ w2t, _Float16* __restrict__ w3t,
                 _Float16* __restrict__ B1t, _Float16* __restrict__ B2t,
                 _Float16* __restrict__ B3t, float* __restrict__ vol) {
  int idx = blockIdx.x * 256 + threadIdx.x;
  if (idx < 73728) {                       // w2t[n][khkw][ci] <- w2[n][ci][khkw]
    int n = idx / 576, r = idx - n * 576;
    int khkw = r >> 6, ci = r & 63;
    w2t[idx] = (_Float16)w2[n * 576 + ci * 9 + khkw];
    return;
  }
  idx -= 73728;
  if (idx < 589824) {                      // w3t[n][khkw][ci] <- w3[n][ci][khkw]
    int n = idx / 1152, r = idx - n * 1152;
    int khkw = r >> 7, ci = r & 127;
    w3t[idx] = (_Float16)w3[n * 1152 + ci * 9 + khkw];
    return;
  }
  idx -= 589824;
  if (idx < 2048) {                        // B1t[64][32], k>=16 zero
    int n = idx >> 5, k = idx & 31;
    B1t[idx] = (k < 16) ? (_Float16)mw1[k * 64 + n] : (_Float16)0.f;
    return;
  }
  idx -= 2048;
  if (idx < 4096) {                        // B2t[64][64]
    int n = idx >> 6, k = idx & 63;
    B2t[idx] = (_Float16)mw2[k * 64 + n];
    return;
  }
  idx -= 4096;
  if (idx < 1024) {                        // B3t[16][64], n>=4 zero
    int n = idx >> 6, k = idx & 63;
    B3t[idx] = (n < 4) ? (_Float16)mw3[k * 4 + n] : (_Float16)0.f;
    return;
  }
  idx -= 1024;
  vol[idx] = 0.f;                          // 16*32*32*32 accumulator
}

// ---------------------------------------------------------------------------
// Kernel 1: conv1 direct (3->64, 3x3, stride2, SAME pad_lo=0) + ReLU.
// Output NHWC f16: x1[v][ho][wo][co].  w1 cached in LDS.
// ---------------------------------------------------------------------------
__global__ __launch_bounds__(256)
void conv1_kernel(const float* __restrict__ img, const float* __restrict__ w1,
                  const float* __restrict__ b1, _Float16* __restrict__ x1) {
  __shared__ float w1s[1728];
  for (int i = threadIdx.x; i < 1728; i += 256) w1s[i] = w1[i];
  __syncthreads();

  int idx = blockIdx.x * 256 + threadIdx.x;   // = ((vi*64+ho)*64+wo)*64+co
  int co = idx & 63;
  int wo = (idx >> 6) & 63;
  int ho = (idx >> 12) & 63;
  int vi = idx >> 18;
  float acc = b1[co];
#pragma unroll
  for (int kh = 0; kh < 3; ++kh) {
    int ih = ho * 2 + kh;
    if (ih >= 128) continue;
#pragma unroll
    for (int kw = 0; kw < 3; ++kw) {
      int iw = wo * 2 + kw;
      if (iw >= 128) continue;
#pragma unroll
      for (int ci = 0; ci < 3; ++ci) {
        float wv = w1s[((co * 3 + ci) * 3 + kh) * 3 + kw];
        float xv = img[(((size_t)vi * 3 + ci) << 14) + (ih << 7) + iw];
        acc = fmaf(wv, xv, acc);
      }
    }
  }
  x1[idx] = (_Float16)fmaxf(acc, 0.f);      // NHWC: flat == idx
}

// ---------------------------------------------------------------------------
// Kernel 2: implicit-GEMM conv via WMMA f16->f32, NHWC activations.
//   out[M=50*32*32][N] = A x W ; K ordered (kh,kw,ci) -> no div/mod, all
//   staging is 64/128-bit vector loads. Block 256 thr (8 waves), tile 32x64.
// ---------------------------------------------------------------------------
__global__ __launch_bounds__(256)
void convgemm_wmma(const _Float16* __restrict__ X, const _Float16* __restrict__ Wt,
                   const float* __restrict__ bias,
                   _Float16* __restrict__ OutAct, float* __restrict__ OutVol,
                   int Cin, int Hin, int Win, int stride, int pad,
                   int Nout, int Ktot, int nBlocksN, float volScale) {
  __shared__ _Float16 As[32][40];   // [m][k-chunk]  (rows 80B, 16B-aligned)
  __shared__ _Float16 Bs[64][40];   // [n][k-chunk]  transposed B tile

  const int tid  = threadIdx.x;
  const int lane = tid & 31;
  const int wv   = tid >> 5;
  const int mi   = wv >> 2;         // M sub-tile 0..1
  const int ni   = wv & 3;          // N sub-tile 0..3
  const int lr   = lane & 15;
  const int hi   = (lane >> 4) & 1;
  const int blkM = (int)blockIdx.x / nBlocksN;
  const int blkN = (int)blockIdx.x % nBlocksN;

  // per-thread A-staging coords: 32 rows x (8 thr/row x 4 halves)
  const int arow_l = tid >> 3;
  const int coff   = (tid & 7) * 4;
  const int m_s    = blkM * 32 + arow_l;
  const int vi_s   = m_s >> 10;               // 32*32 outputs per view
  const int oh_s   = (m_s >> 5) & 31;
  const int ow_s   = m_s & 31;
  // per-thread B-staging coords: 64 rows x (4 thr/row x 8 halves)
  const int brow_l = tid >> 2;
  const int koff_b = (tid & 3) * 8;
  const _Float16* wrow = Wt + (size_t)(blkN * 64 + brow_l) * Ktot;

  v8f acc = {};

  for (int kh = 0; kh < 3; ++kh) {
    const int ih = oh_s * stride + kh - pad;
    const bool vh = (unsigned)ih < (unsigned)Hin;
    for (int kw = 0; kw < 3; ++kw) {
      const int iw = ow_s * stride + kw - pad;
      const bool vrow = vh && ((unsigned)iw < (unsigned)Win);
      const _Float16* arow =
          X + (((size_t)vi_s * Hin + ih) * Win + iw) * Cin;
      const int kbase = (kh * 3 + kw) * Cin;
      for (int c0 = 0; c0 < Cin; c0 += 32) {
        __syncthreads();
        // ---- stage A (b64 vector load, zero-fill at borders) ----
        v4h av = {};
        if (vrow) av = *(const v4h*)(arow + c0 + coff);
        *(v4h*)&As[arow_l][coff] = av;
        // ---- stage B transposed (b128 vector load) ----
        *(v8h*)&Bs[brow_l][koff_b] = *(const v8h*)(wrow + kbase + c0 + koff_b);
        if (kbase + c0 + 32 < Ktot)   // prefetch next weight K-slice
          __builtin_prefetch(wrow + kbase + c0 + 32 + koff_b, 0, 1);
        __syncthreads();

        // ---- fragments: CDNA5 16x16x32 f16 layout, all 16B LDS loads ----
        const _Float16* ar = &As[mi * 16 + lr][0];
        v16h a = cat8(*(const v8h*)(ar + hi * 8),
                      *(const v8h*)(ar + 16 + hi * 8));
        const _Float16* br = &Bs[ni * 16 + lr][0];
        v16h b = cat8(*(const v8h*)(br + hi * 8),
                      *(const v8h*)(br + 16 + hi * 8));
        acc = __builtin_amdgcn_wmma_f32_16x16x32_f16(false, a, false, b,
                                                     (short)0, acc, false, false);
      }
    }
  }

  // ---- epilogue: C layout M = r + 8*hi, N = lane&15 ----
  const int n     = blkN * 64 + ni * 16 + lr;
  const float bv  = bias[n];
  const int mbase = blkM * 32 + mi * 16 + hi * 8;
#pragma unroll
  for (int r = 0; r < 8; ++r) {
    float v = acc[r] + bv;
    int m   = mbase + r;
    if (OutAct) {                         // NHWC: flat = m*Nout + n
      OutAct[(size_t)m * Nout + n] = (_Float16)fmaxf(v, 0.f);
    } else {                              // vol flat index == n*1024 + hw
      atomicAdd(&OutVol[((size_t)n << 10) + (m & 1023)], v * volScale);
    }
  }
}

// ---------------------------------------------------------------------------
// Kernel 3: ray sampling + trilinear interpolation -> feat[NPTS][16] f16
// ---------------------------------------------------------------------------
__global__ __launch_bounds__(256)
void sample_kernel(const float* __restrict__ rays, const float* __restrict__ vol,
                   _Float16* __restrict__ feat) {
  int p  = blockIdx.x * 256 + threadIdx.x;   // p = s*4096 + h*64 + w
  int s  = p >> 12;
  int hw = p & 4095;
  float t  = (float)s * (1.f / (NS - 1));
  float nv = rays[6 * 4096 + hw];
  float fv = rays[7 * 4096 + hw];
  float z  = nv + (fv - nv) * t;
  float cc[3];
#pragma unroll
  for (int c = 0; c < 3; ++c) {
    float pt = rays[c * 4096 + hw] + rays[(3 + c) * 4096 + hw] * z;
    cc[c] = fminf(fmaxf((pt + 1.f) * 0.5f * (DD - 1), 0.f), (float)(DD - 1));
  }
  int ix = min((int)floorf(cc[0]), DD - 2);
  int iy = min((int)floorf(cc[1]), DD - 2);
  int iz = min((int)floorf(cc[2]), DD - 2);
  float fx = cc[0] - ix, fy = cc[1] - iy, fz = cc[2] - iz;

  int   off[8];
  float wc [8];
#pragma unroll
  for (int c8 = 0; c8 < 8; ++c8) {
    int dz = (c8 >> 2) & 1, dy = (c8 >> 1) & 1, dx = c8 & 1;
    off[c8] = (iz + dz) * 1024 + (iy + dy) * 32 + (ix + dx);
    wc[c8]  = (dz ? fz : 1.f - fz) * (dy ? fy : 1.f - fy) * (dx ? fx : 1.f - fx);
  }
  v8h f0, f1;
#pragma unroll
  for (int ch = 0; ch < CCH; ++ch) {
    const float* vb = vol + ((size_t)ch << 15);
    float a = 0.f;
#pragma unroll
    for (int c8 = 0; c8 < 8; ++c8) a = fmaf(wc[c8], vb[off[c8]], a);
    if (ch < 8) f0[ch] = (_Float16)a; else f1[ch - 8] = (_Float16)a;
  }
  v8h* dst = (v8h*)(feat + (size_t)p * 16);
  dst[0] = f0;
  dst[1] = f1;
}

// ---------------------------------------------------------------------------
// Kernel 4: fused 16->64->64->4 MLP, fully WMMA. One wave handles 16 points.
// All weight matrices stored transposed [n][k] -> contiguous b-fragments.
// ---------------------------------------------------------------------------
__global__ __launch_bounds__(256)
void mlp_wmma_kernel(const _Float16* __restrict__ feat,
                     const _Float16* __restrict__ B1t, const _Float16* __restrict__ B2t,
                     const _Float16* __restrict__ B3t,
                     const float* __restrict__ mb1, const float* __restrict__ mb2,
                     const float* __restrict__ mb3, float* __restrict__ raw) {
  __shared__ _Float16 hb[8][16][72];          // per-wave 16x64 hidden (+pad)

  const int tid  = threadIdx.x;
  const int lane = tid & 31;
  const int wv   = tid >> 5;
  const int lr   = lane & 15;
  const int hi   = (lane >> 4) & 1;
  const int p0   = ((int)blockIdx.x * 8 + wv) * 16;

  // --- A1 fragment: feat row (K=16 real, upper half zero-pad) ---
  const v8h zz = {};
  v16h a1 = cat8(*(const v8h*)(feat + (size_t)(p0 + lr) * 16 + hi * 8), zz);

  // --- layer 1: (16x32) x (32x64) in 4 N-tiles ---
#pragma unroll
  for (int jn = 0; jn < 4; ++jn) {
    const _Float16* br = B1t + (jn * 16 + lr) * 32;
    v16h b = cat8(*(const v8h*)(br + hi * 8), *(const v8h*)(br + 16 + hi * 8));
    v8f acc = {};
    acc = __builtin_amdgcn_wmma_f32_16x16x32_f16(false, a1, false, b,
                                                 (short)0, acc, false, false);
    float bv = mb1[jn * 16 + lr];
#pragma unroll
    for (int r = 0; r < 8; ++r)
      hb[wv][r + hi * 8][jn * 16 + lr] = (_Float16)fmaxf(acc[r] + bv, 0.f);
  }
  __syncthreads();

  // --- layer 2: (16x64) x (64x64): 2 K-chunks x 4 N-tiles ---
  v16h a2[2];
#pragma unroll
  for (int tc = 0; tc < 2; ++tc) {
    const _Float16* hr = &hb[wv][lr][tc * 32];
    a2[tc] = cat8(*(const v8h*)(hr + hi * 8), *(const v8h*)(hr + 16 + hi * 8));
  }
  __syncthreads();

#pragma unroll
  for (int jn = 0; jn < 4; ++jn) {
    v8f acc = {};
#pragma unroll
    for (int tc = 0; tc < 2; ++tc) {
      const _Float16* br = B2t + (jn * 16 + lr) * 64 + tc * 32;
      v16h b = cat8(*(const v8h*)(br + hi * 8), *(const v8h*)(br + 16 + hi * 8));
      acc = __builtin_amdgcn_wmma_f32_16x16x32_f16(false, a2[tc], false, b,
                                                   (short)0, acc, false, false);
    }
    float bv = mb2[jn * 16 + lr];
#pragma unroll
    for (int r = 0; r < 8; ++r)
      hb[wv][r + hi * 8][jn * 16 + lr] = (_Float16)fmaxf(acc[r] + bv, 0.f);
  }
  __syncthreads();

  // --- layer 3: (16x64) x (64x16), first 4 cols real ---
  v16h a3[2];
#pragma unroll
  for (int tc = 0; tc < 2; ++tc) {
    const _Float16* hr = &hb[wv][lr][tc * 32];
    a3[tc] = cat8(*(const v8h*)(hr + hi * 8), *(const v8h*)(hr + 16 + hi * 8));
  }
  v8f acc = {};
#pragma unroll
  for (int tc = 0; tc < 2; ++tc) {
    const _Float16* br = B3t + lr * 64 + tc * 32;
    v16h b = cat8(*(const v8h*)(br + hi * 8), *(const v8h*)(br + 16 + hi * 8));
    acc = __builtin_amdgcn_wmma_f32_16x16x32_f16(false, a3[tc], false, b,
                                                 (short)0, acc, false, false);
  }
  if (lr < 4) {
    float bv = mb3[lr];
#pragma unroll
    for (int r = 0; r < 8; ++r)
      raw[(size_t)(p0 + r + hi * 8) * 4 + lr] = acc[r] + bv;
  }
}

// ---------------------------------------------------------------------------
// Kernel 5: alpha compositing along S, one thread per ray (float4 loads).
// ---------------------------------------------------------------------------
__global__ __launch_bounds__(256)
void composite_kernel(const float* __restrict__ rays, const float* __restrict__ raw,
                      float* __restrict__ out) {
  int ray = blockIdx.x * 256 + threadIdx.x;   // 0..4095
  float nv = rays[6 * 4096 + ray];
  float fv = rays[7 * 4096 + ray];
  float dz = (fv - nv) * (1.f / (NS - 1));
  float T = 1.f, aR = 0.f, aG = 0.f, aB = 0.f, aD = 0.f;
  const float4* rp4 = (const float4*)raw;
  for (int s = 0; s < NS; ++s) {
    float4 rp = rp4[(s << 12) + ray];
    float sigma = fmaxf(rp.w, 0.f);
    float z     = nv + dz * (float)s;
    float delta = (s == NS - 1) ? 1e10f : dz;
    float alpha = 1.f - __expf(-sigma * delta);
    float w     = alpha * T;
    aR += w / (1.f + __expf(-rp.x));
    aG += w / (1.f + __expf(-rp.y));
    aB += w / (1.f + __expf(-rp.z));
    aD += w * z;
    T *= (1.f - alpha + 1e-10f);
  }
  out[0 * 4096 + ray] = aR;
  out[1 * 4096 + ray] = aG;
  out[2 * 4096 + ray] = aB;
  out[3 * 4096 + ray] = aD;
}

// ---------------------------------------------------------------------------
extern "C" void kernel_launch(void* const* d_in, const int* in_sizes, int n_in,
                              void* d_out, int out_size, void* d_ws, size_t ws_size,
                              hipStream_t stream) {
  const float* images = (const float*)d_in[0];
  const float* rays   = (const float*)d_in[1];
  const float* w1     = (const float*)d_in[2];
  const float* b1     = (const float*)d_in[3];
  const float* w2     = (const float*)d_in[4];
  const float* b2     = (const float*)d_in[5];
  const float* w3     = (const float*)d_in[6];
  const float* b3     = (const float*)d_in[7];
  const float* mw1    = (const float*)d_in[8];
  const float* mb1    = (const float*)d_in[9];
  const float* mw2    = (const float*)d_in[10];
  const float* mb2    = (const float*)d_in[11];
  const float* mw3    = (const float*)d_in[12];
  const float* mb3    = (const float*)d_in[13];

  char* ws = (char*)d_ws;
  size_t off = 0;
  _Float16* x1   = (_Float16*)(ws + off); off += (size_t)NVIEW * 64 * 64 * C1 * 2; // NHWC, 26.2 MB
  _Float16* x2   = (_Float16*)(ws + off); off += (size_t)NVIEW * 32 * 32 * C2 * 2; // NHWC, 13.1 MB
  _Float16* w2t  = (_Float16*)(ws + off); off += (size_t)73728 * 2;
  _Float16* w3t  = (_Float16*)(ws + off); off += (size_t)589824 * 2;
  _Float16* B1t  = (_Float16*)(ws + off); off += (size_t)2048 * 2;
  _Float16* B2t  = (_Float16*)(ws + off); off += (size_t)4096 * 2;
  _Float16* B3t  = (_Float16*)(ws + off); off += (size_t)1024 * 2;
  float*    vol  = (float*)(ws + off);    off += (size_t)CCH * DD * DD * DD * 4;   // 2 MB
  _Float16* feat = (_Float16*)(ws + off); off += (size_t)NPTS * 16 * 2;            // 8.4 MB
  float*    raw  = (float*)(ws + off);    off += (size_t)NPTS * 4 * 4;             // 4.2 MB
  (void)ws_size; (void)n_in; (void)in_sizes; (void)out_size;

  // 0) weight transposition -> f16, zero mean-volume
  prep_kernel<<<4668, 256, 0, stream>>>(w2, w3, mw1, mw2, mw3,
                                        w2t, w3t, B1t, B2t, B3t, vol);
  // 1) conv1 direct + ReLU -> x1 (NHWC f16)
  conv1_kernel<<<51200, 256, 0, stream>>>(images, w1, b1, x1);
  // 2) conv2: M=51200 N=128 K=576, stride2 pad0, ReLU -> x2 (NHWC)
  convgemm_wmma<<<1600 * 2, 256, 0, stream>>>(x1, w2t, b2, x2, nullptr,
                                              C1, 64, 64, 2, 0, C2, 576, 2, 0.f);
  // 3) conv3: M=51200 N=512 K=1152, stride1 pad1, mean over views -> vol
  convgemm_wmma<<<1600 * 8, 256, 0, stream>>>(x2, w3t, b3, nullptr, vol,
                                              C2, 32, 32, 1, 1, C3, 1152, 8,
                                              1.f / (float)NVIEW);
  // 4) ray sampling + trilinear -> feat f16
  sample_kernel<<<NPTS / 256, 256, 0, stream>>>(rays, vol, feat);
  // 5) fused WMMA MLP -> raw (S,H,W,4)
  mlp_wmma_kernel<<<NPTS / 128, 256, 0, stream>>>(feat, B1t, B2t, B3t,
                                                  mb1, mb2, mb3, raw);
  // 6) alpha compositing -> d_out (3*4096 rgb planes + 4096 depth)
  composite_kernel<<<16, 256, 0, stream>>>(rays, raw, (float*)d_out);
}
